// NestedConv_3496103379078
// MI455X (gfx1250) — compile-verified, matrix operand order checked by reading, not run on hardware
//
#include <hip/hip_runtime.h>

// NestedConv fused kernel for MI455X (gfx1250, wave32, WMMA + TDM).
//
// out[b,i,j,d] = mask[b,i,j] * ( A[b] @ relu(relu(mask*X @ W1 + b1) @ W2 + b2) )[j,d]
// (A symmetric, so einsum 'bikd,bkj->bijd' == A[b] (128x128) @ h[b,i] (128x64)).
//
// One workgroup (256 thr = 8 wave32) per (b,i).
//  - X tile (32 KB f32) DMA'd into LDS staging via tensor_load_to_lds (TENSORcnt)
//  - masked f32->f16 conversion in LDS, W1/W2 transposed in LDS, A[b] in LDS f16
//  - three chained v_wmma_f32_16x16x32_f16 matmuls with f32 accumulation
// Memory-bound: ~517 MB HBM traffic -> ~22 us floor; 34 GF of f16 WMMA is far
// below the f16 matrix-op ceiling, so data movement dominates as intended.

#define BB 64
#define NN 128
#define DD 64

typedef unsigned int u32;
typedef __attribute__((ext_vector_type(4)))  u32      u32x4;
typedef __attribute__((ext_vector_type(8)))  u32      u32x8;
typedef __attribute__((ext_vector_type(16))) _Float16 v16h;
typedef __attribute__((ext_vector_type(8)))  _Float16 v8h;
typedef __attribute__((ext_vector_type(8)))  float    v8f;

union Frag16 { v16h v; struct { v8h lo; v8h hi; } p; };

// Load a 16x32 f16 WMMA operand fragment from an LDS matrix whose K-dimension
// is contiguous (stride in halves between consecutive "major" rows/cols).
// CDNA5 16-bit operand striping:
//   lanes 0-15 : major = lane,    elems 0-7 = K+0..7,   elems 8-15 = K+16..23
//   lanes16-31 : major = lane-16, elems 0-7 = K+8..15,  elems 8-15 = K+24..31
__device__ __forceinline__ v16h load_frag(const _Float16* __restrict__ base,
                                          int majorBase, int kBase, int stride,
                                          int lane) {
  const int m15 = lane & 15;
  const int hs  = (lane >> 4) & 1;
  const _Float16* p = base + (majorBase + m15) * stride + kBase + hs * 8;
  Frag16 f;
  f.p.lo = *reinterpret_cast<const v8h*>(p);        // ds_load_b128
  f.p.hi = *reinterpret_cast<const v8h*>(p + 16);   // ds_load_b128
  return f.v;
}

__global__ __launch_bounds__(256)
void nestedconv_fused_kernel(const float* __restrict__ X,
                             const float* __restrict__ A,
                             const unsigned char* __restrict__ mask,
                             const float* __restrict__ W1,
                             const float* __restrict__ b1,
                             const float* __restrict__ W2,
                             const float* __restrict__ b2,
                             float* __restrict__ out) {
  // 128 KB LDS total -> 2 workgroups / WGP of the 320 KB pool.
  __shared__ __align__(16) float    sXstage[NN * DD]; // 32 KB raw f32 TDM landing pad
  __shared__ __align__(16) _Float16 sX  [NN * DD];    // masked X tile f16, row-major (k,d)
  __shared__ __align__(16) _Float16 sW1t[DD * DD];    // W1 transposed: (n,k)
  __shared__ __align__(16) _Float16 sW2t[DD * DD];    // W2 transposed: (n,k)
  __shared__ __align__(16) _Float16 sT1 [NN * DD];    // relu(Xm@W1+b1), row-major (k,d)
  __shared__ __align__(16) _Float16 sT2t[DD * NN];    // relu(T1@W2+b2) transposed: (d,k)
  __shared__ __align__(16) _Float16 sA  [NN * NN];    // A[b] f16, row-major (j,k)

  const int bi = blockIdx.x;            // bi = b*NN + i
  const int b  = bi >> 7;
  const int t  = threadIdx.x;           // 0..255
  const int wave = t >> 5;              // 0..7
  const int lane = t & 31;

  const float*         Xt = X    + (size_t)bi * (NN * DD);
  const unsigned char* mr = mask + (size_t)bi * NN;
  const float*         Ab = A    + (size_t)b  * (NN * NN);

  // ---- Stage 0a: kick off TDM DMA of the raw X tile into LDS staging -------
  // D# per CDNA5 ISA: 1 valid descriptor, 2D degenerate
  // (single row of 8192 4-byte elements), groups 2/3 present-but-unused.
  if (wave == 0) {
    const unsigned long long ga  = (unsigned long long)(const void*)Xt;
    const u32 ldsOff = (u32)(size_t)(void*)sXstage;  // LDS aperture: low 32 bits
    u32x4 g0;
    g0[0] = 1u;                                   // count=1, is_restore=0, gather=0
    g0[1] = ldsOff;                               // lds_addr (bytes)
    g0[2] = (u32)(ga & 0xffffffffu);              // global_addr[31:0]
    g0[3] = (u32)((ga >> 32) & 0x01ffffffu)       // global_addr[56:32]
            | 0x80000000u;                        // type=2 ("image")
    u32x8 g1;
    g1[0] = 0x00020000u;                          // wg_mask=0, data_size=2 (4B)
    g1[1] = (8192u & 0xffffu) << 16;              // tensor_dim0[15:0] = 8192
    g1[2] = (1u << 16);                           // dim0 hi=0 | tensor_dim1=1
    g1[3] = (8192u & 0xffffu) << 16;              // dim1 hi=0 | tile_dim0=8192
    g1[4] = 1u;                                   // tile_dim1=1, tile_dim2=0
    g1[5] = 8192u;                                // tensor_dim0_stride[31:0]
    g1[6] = 0u;                                   // stride0 hi | stride1 lo
    g1[7] = 0u;                                   // stride1 hi
    u32x4 g2 = {0u, 0u, 0u, 0u};                  // tensor_dim2/3, tile_dim3 unused
    u32x4 g3 = {0u, 0u, 0u, 0u};                  // tensor_dim3_stride/dim4 unused
    asm volatile("tensor_load_to_lds %0, %1, %2, %3"
                 :
                 : "s"(g0), "s"(g1), "s"(g2), "s"(g3)
                 : "memory");
  }

  // ---- Stage 0b: W1/W2/A global -> LDS (f32->f16) while the DMA runs -------
  #pragma unroll
  for (int it = 0; it < 16; ++it) {
    const int e = t + it * 256;            // 0..4095
    const int k = e >> 6;
    const int n = e & 63;
    sW1t[n * DD + k] = (_Float16)W1[e];    // transposed: B-frag contiguous in K
    sW2t[n * DD + k] = (_Float16)W2[e];
  }
  #pragma unroll
  for (int it = 0; it < 16; ++it) {
    const int v4 = t + it * 256;
    const float4 a4 = reinterpret_cast<const float4*>(Ab)[v4];
    const int e = v4 * 4;
    sA[e + 0] = (_Float16)a4.x;
    sA[e + 1] = (_Float16)a4.y;
    sA[e + 2] = (_Float16)a4.z;
    sA[e + 3] = (_Float16)a4.w;
  }
  // Prefetch next block's X tile toward L2 (global_prefetch_b8 path).
  if (bi + 1 < (int)gridDim.x) {
    __builtin_prefetch(reinterpret_cast<const char*>(Xt + NN * DD) + t * 128, 0, 1);
  }

  // ---- Stage 0c: drain TENSORcnt, publish staging, masked convert ----------
  if (wave == 0) {
    __builtin_amdgcn_s_wait_tensorcnt(0);  // s_wait_tensorcnt 0x0
  }
  __syncthreads();

  #pragma unroll
  for (int it = 0; it < 8; ++it) {
    const int v4 = t + it * 256;           // float4 index into staged tile
    const int e  = v4 * 4;
    const int k  = e >> 6;                 // row (d-stride 64)
    const float m = mr[k] ? 1.0f : 0.0f;
    const float4 x = reinterpret_cast<const float4*>(sXstage)[v4];
    sX[e + 0] = (_Float16)(x.x * m);
    sX[e + 1] = (_Float16)(x.y * m);
    sX[e + 2] = (_Float16)(x.z * m);
    sX[e + 3] = (_Float16)(x.w * m);
  }
  __syncthreads();

  const int rowBase = wave * 16;            // 16 output rows per wave
  const int n15 = lane & 15;
  const int mo  = (lane >> 4) * 8;          // C/D layout: lanes 16-31 hold M+8

  // ---- Stage 1: T1 = relu(Xm @ W1 + b1)  (128x64 @ 64x64) ------------------
  {
    v8f acc[4] = {};
    #pragma unroll
    for (int kc = 0; kc < DD; kc += 32) {
      const v16h af = load_frag(sX, rowBase, kc, DD, lane);
      #pragma unroll
      for (int nt = 0; nt < 4; ++nt) {
        const v16h bf = load_frag(sW1t, nt * 16, kc, DD, lane);
        acc[nt] = __builtin_amdgcn_wmma_f32_16x16x32_f16(
            false, af, false, bf, (short)0, acc[nt], false, false);
      }
    }
    #pragma unroll
    for (int nt = 0; nt < 4; ++nt) {
      const float bias = b1[nt * 16 + n15];
      #pragma unroll
      for (int r = 0; r < 8; ++r) {
        float v = acc[nt][r] + bias;
        v = v > 0.0f ? v : 0.0f;
        sT1[(rowBase + mo + r) * DD + nt * 16 + n15] = (_Float16)v;
      }
    }
  }
  __syncthreads();

  // ---- Stage 2: T2 = relu(T1 @ W2 + b2), stored transposed (d,k) -----------
  {
    v8f acc[4] = {};
    #pragma unroll
    for (int kc = 0; kc < DD; kc += 32) {
      const v16h af = load_frag(sT1, rowBase, kc, DD, lane);
      #pragma unroll
      for (int nt = 0; nt < 4; ++nt) {
        const v16h bf = load_frag(sW2t, nt * 16, kc, DD, lane);
        acc[nt] = __builtin_amdgcn_wmma_f32_16x16x32_f16(
            false, af, false, bf, (short)0, acc[nt], false, false);
      }
    }
    #pragma unroll
    for (int nt = 0; nt < 4; ++nt) {
      const float bias = b2[nt * 16 + n15];
      #pragma unroll
      for (int r = 0; r < 8; ++r) {
        float v = acc[nt][r] + bias;
        v = v > 0.0f ? v : 0.0f;
        sT2t[(nt * 16 + n15) * NN + rowBase + mo + r] = (_Float16)v;
      }
    }
  }
  __syncthreads();

  // ---- Stage 3: Out = A[b] @ T2  (128x128 @ 128x64), mask rows, store f32 --
  {
    v8f acc[4] = {};
    #pragma unroll
    for (int kc = 0; kc < NN; kc += 32) {
      const v16h af = load_frag(sA, rowBase, kc, NN, lane);
      #pragma unroll
      for (int nt = 0; nt < 4; ++nt) {
        const v16h bf = load_frag(sT2t, nt * 16, kc, NN, lane);
        acc[nt] = __builtin_amdgcn_wmma_f32_16x16x32_f16(
            false, af, false, bf, (short)0, acc[nt], false, false);
      }
    }
    float* op = out + (size_t)bi * (NN * DD);
    #pragma unroll
    for (int nt = 0; nt < 4; ++nt) {
      #pragma unroll
      for (int r = 0; r < 8; ++r) {
        const int j = rowBase + mo + r;
        const float v = mr[j] ? acc[nt][r] : 0.0f;
        op[j * DD + nt * 16 + n15] = v;
      }
    }
  }
}

extern "C" void kernel_launch(void* const* d_in, const int* in_sizes, int n_in,
                              void* d_out, int out_size, void* d_ws, size_t ws_size,
                              hipStream_t stream) {
  (void)in_sizes; (void)n_in; (void)out_size; (void)d_ws; (void)ws_size;
  const float*         X    = (const float*)d_in[0];
  const float*         A    = (const float*)d_in[1];
  const unsigned char* mask = (const unsigned char*)d_in[2];  // jax bool = 1 byte/elem
  const float*         W1   = (const float*)d_in[3];
  const float*         b1   = (const float*)d_in[4];
  const float*         W2   = (const float*)d_in[5];
  const float*         b2   = (const float*)d_in[6];
  float*               out  = (float*)d_out;

  dim3 grid(BB * NN);   // one workgroup per (b, i)
  dim3 block(256);      // 8 wave32s
  hipLaunchKernelGGL(nestedconv_fused_kernel, grid, block, 0, stream,
                     X, A, mask, W1, b1, W2, b2, out);
}
